// Model_53257594470527
// MI455X (gfx1250) — matrix operand order repaired and checked by reading, not compiled, not run on hardware
//
#include <hip/hip_runtime.h>

typedef __attribute__((ext_vector_type(16))) __bf16 v16bf;
typedef __attribute__((ext_vector_type(8)))  float  v8f;

#define NN    50000
#define EN    800000
#define EEDG  850000        // EN + NN self loops
#define HID   128
#define HF    152
#define XP    160           // padded xp row stride
#define HEADS 8
#define OUTC  19
#define SLOPE 0.2f
#define EPSV  1e-16f

// ---------------- embedding gather ----------------
__global__ void k_embed(const int* __restrict__ z, const float* __restrict__ emb,
                        float* __restrict__ h) {
  int i = blockIdx.x * blockDim.x + threadIdx.x;
  if (i < NN * HID) {
    int n = i / HID, c = i % HID;
    h[i] = emb[z[n] * HID + c];
  }
}

// ---------------- zero scratch ----------------
__global__ void k_zero(float* __restrict__ p, int cnt) {
  int i = blockIdx.x * blockDim.x + threadIdx.x;
  if (i < cnt) p[i] = 0.0f;
}

// Fragment-layout helpers (ISA 7.12.2, wave32):
//  A (16x32, 16-bit): lane = g*16 + m,  K = j + 16*g
//  B (32x16, 16-bit): lane = g*16 + n,  K = j + 16*g
// LDS holds fragments pre-swizzled: each lane's 16 halfs contiguous (32B).

// ---- pack weights (f32 global) -> fragment-ordered bf16 image (global) ----
__global__ void k_packB1(const float* __restrict__ W, __bf16* __restrict__ Bpk) {
  int idx = blockIdx.x * blockDim.x + threadIdx.x;
  if (idx >= HID * 160) return;
  int k = idx / 160, n = idx % 160;
  float val = (n < HF) ? W[k * HF + n] : 0.0f;
  int nT = n >> 4, l15 = n & 15;
  int kt = k >> 5, r = k & 31;
  int g = r >> 4, j = r & 15;
  Bpk[(((nT * 4) + kt) * 32 + (g * 16 + l15)) * 16 + j] = (__bf16)val;
}

__global__ void k_packB2(const float* __restrict__ fW, __bf16* __restrict__ Bpk) {
  int idx = blockIdx.x * blockDim.x + threadIdx.x;
  if (idx >= 160 * HID) return;
  int k = idx / HID, n = idx % HID;
  float val = (k < HF) ? fW[k * HID + n] : 0.0f;
  int nT = n >> 4, l15 = n & 15;
  int kt = k >> 5, r = k & 31;
  int g = r >> 4, j = r & 15;
  Bpk[(((nT * 5) + kt) * 32 + (g * 16 + l15)) * 16 + j] = (__bf16)val;
}

// async byte-move global -> LDS (CDNA5, ASYNCcnt tracked)
__device__ __forceinline__ void async_lds_b128(unsigned lds_addr, unsigned voff,
                                               const void* saddr) {
  asm volatile("global_load_async_to_lds_b128 %0, %1, %2"
               :: "v"(lds_addr), "v"(voff), "s"(saddr) : "memory");
}
__device__ __forceinline__ void wait_async0() {
  asm volatile("s_wait_asynccnt 0x0" ::: "memory");
}
__device__ __forceinline__ unsigned lds_off32(const void* p) {
  // generic pointer to LDS: aperture in [63:32], LDS byte address in [31:0]
  return (unsigned)(unsigned long long)p;
}

// ---------------- GEMM1: xp[N,160pad] = h[N,128] @ W[128,152] (WMMA bf16) ---
// 320 threads = 10 waves; wave w owns N-tile w; block covers 32 node rows
// (2 M-tiles per wave, B fragment reused). xp padded to 160 -> no col guard.
__global__ __launch_bounds__(320) void k_gemm1(const float* __restrict__ h,
                                               const __bf16* __restrict__ Bpk,
                                               float* __restrict__ xp) {
  __shared__ __attribute__((aligned(32))) __bf16 Ap[2 * 4 * 32 * 16]; //  8 KB
  __shared__ __attribute__((aligned(32))) __bf16 Bp[10 * 4 * 32 * 16]; // 40 KB
  const int tid = threadIdx.x;
  const int m0  = blockIdx.x * 32;

  // async-copy pre-packed B fragments: 40960 B = 2560 x b128
  const unsigned ldsB = lds_off32(&Bp[0]);
  for (int c = tid; c < 2560; c += 320) {
    unsigned off = (unsigned)c * 16u;
    async_lds_b128(ldsB + off, off, Bpk);
  }
  // stage h rows -> packed A fragments (f32 -> bf16 conversion)
  for (int idx = tid; idx < 32 * HID; idx += 320) {
    int m32 = idx / HID, k = idx % HID;
    int node = m0 + m32;
    float val = (node < NN) ? h[node * HID + k] : 0.0f;
    int mt = m32 >> 4, m = m32 & 15;
    int kt = k >> 5, r = k & 31;
    int g = r >> 4;
    int j = r & 15;          // A map
    Ap[(((mt * 4) + kt) * 32 + (g * 16 + m)) * 16 + j] = (__bf16)val;
  }
  wait_async0();
  __syncthreads();

  const int lane = tid & 31;
  const int wave = tid >> 5;
  const int g    = lane >> 4;
  const int l15  = lane & 15;

  v8f acc0 = {}, acc1 = {};
  for (int kt = 0; kt < 4; ++kt) {
    v16bf b  = *(const v16bf*)&Bp[((wave * 4 + kt) * 32 + lane) * 16];
    v16bf a0 = *(const v16bf*)&Ap[((0 * 4 + kt) * 32 + lane) * 16];
    v16bf a1 = *(const v16bf*)&Ap[((1 * 4 + kt) * 32 + lane) * 16];
    acc0 = __builtin_amdgcn_wmma_f32_16x16x32_bf16(false, a0, false, b,
                                                   (short)0, acc0, false, false);
    acc1 = __builtin_amdgcn_wmma_f32_16x16x32_bf16(false, a1, false, b,
                                                   (short)0, acc1, false, false);
  }
  const int n = wave * 16 + l15;
  const int row0 = m0 + g * 8;
  if (m0 + 32 <= NN) {            // block-uniform fast path
#pragma unroll
    for (int r = 0; r < 8; ++r) {
      xp[(row0 + r) * XP + n]      = acc0[r];
      xp[(row0 + r + 16) * XP + n] = acc1[r];
    }
  } else {
#pragma unroll
    for (int r = 0; r < 8; ++r) {
      if (row0 + r < NN)      xp[(row0 + r) * XP + n]      = acc0[r];
      if (row0 + r + 16 < NN) xp[(row0 + r + 16) * XP + n] = acc1[r];
    }
  }
}

// ---------------- per-node, per-head attention dots ----------------
__global__ void k_al(const float* __restrict__ xp, const float* __restrict__ a_s,
                     const float* __restrict__ a_d, float* __restrict__ als,
                     float* __restrict__ ald) {
  int i = blockIdx.x * blockDim.x + threadIdx.x;
  if (i >= NN * HEADS) return;
  int n = i / HEADS, hh = i % HEADS;
  const float* xr = xp + n * XP + hh * OUTC;
  const float* as = a_s + hh * OUTC;
  const float* ad = a_d + hh * OUTC;
  float s = 0.f, d = 0.f;
#pragma unroll
  for (int f = 0; f < OUTC; ++f) { s += xr[f] * as[f]; d += xr[f] * ad[f]; }
  als[i] = s;
  ald[i] = d;
}

// order-preserving float <-> uint for atomicMax-based segment max
__device__ __forceinline__ unsigned f2ord(float f) {
  unsigned b = __float_as_uint(f);
  return (b & 0x80000000u) ? ~b : (b | 0x80000000u);
}
__device__ __forceinline__ float ord2f(unsigned u) {
  unsigned b = (u & 0x80000000u) ? (u ^ 0x80000000u) : ~u;
  return __uint_as_float(b);
}

// ---------------- edge pass 1: logits + segment max ----------------
__global__ void k_edge1(const int* __restrict__ ei, const float* __restrict__ als,
                        const float* __restrict__ ald, float* __restrict__ ebuf,
                        unsigned* __restrict__ maxb) {
  int i = blockIdx.x * blockDim.x + threadIdx.x;
  if (i >= EEDG * HEADS) return;
  int e = i / HEADS, hh = i % HEADS;
  int s, d;
  if (e < EN) { s = ei[e]; d = ei[EN + e]; } else { s = d = e - EN; }
  float v = als[s * HEADS + hh] + ald[d * HEADS + hh];
  v = (v > 0.f) ? v : SLOPE * v;           // leaky_relu
  ebuf[i] = v;
  atomicMax(&maxb[d * HEADS + hh], f2ord(v));
}

// ---------------- edge pass 2: exp + segment sum ----------------
__global__ void k_edge2(const int* __restrict__ ei, float* __restrict__ ebuf,
                        const unsigned* __restrict__ maxb, float* __restrict__ sumb) {
  int i = blockIdx.x * blockDim.x + threadIdx.x;
  if (i >= EEDG * HEADS) return;
  int e = i / HEADS, hh = i % HEADS;
  int d = (e < EN) ? ei[EN + e] : e - EN;
  float m = ord2f(maxb[d * HEADS + hh]);
  float a = __expf(ebuf[i] - m);
  ebuf[i] = a;
  atomicAdd(&sumb[d * HEADS + hh], a);
}

// ---------------- edge pass 2b: normalize alpha in place ----------------
__global__ void k_alpha(const int* __restrict__ ei, float* __restrict__ ebuf,
                        const float* __restrict__ sumb) {
  int i = blockIdx.x * blockDim.x + threadIdx.x;
  if (i >= EEDG * HEADS) return;
  int e = i / HEADS, hh = i % HEADS;
  int d = (e < EN) ? ei[EN + e] : e - EN;
  ebuf[i] = ebuf[i] / (sumb[d * HEADS + hh] + EPSV);
}

// ---------------- edge pass 3: weighted message scatter-add (float4) --------
__global__ void k_edge3(const int* __restrict__ ei, const float* __restrict__ alpha,
                        const float* __restrict__ xp, float* __restrict__ msg) {
  long long i = (long long)blockIdx.x * blockDim.x + threadIdx.x;
  if (i >= (long long)EEDG * (HF / 4)) return;
  int e  = (int)(i / (HF / 4));
  int c0 = 4 * (int)(i % (HF / 4));
  int s, d;
  if (e < EN) { s = ei[e]; d = ei[EN + e]; } else { s = d = e - EN; }
  const float4 xv = *(const float4*)&xp[s * XP + c0];
  const float* al = alpha + e * HEADS;
  float v0 = al[(c0 + 0) / OUTC] * xv.x;
  float v1 = al[(c0 + 1) / OUTC] * xv.y;
  float v2 = al[(c0 + 2) / OUTC] * xv.z;
  float v3 = al[(c0 + 3) / OUTC] * xv.w;
  float* mrow = msg + d * HF + c0;
  atomicAdd(&mrow[0], v0);
  atomicAdd(&mrow[1], v1);
  atomicAdd(&mrow[2], v2);
  atomicAdd(&mrow[3], v3);
}

// ---------------- bias + LayerNorm, in place (wave32 per node) ----------------
__global__ __launch_bounds__(256) void k_post(float* __restrict__ msg,
                                              const float* __restrict__ gb,
                                              const float* __restrict__ gam,
                                              const float* __restrict__ bet) {
  int node = (blockIdx.x * blockDim.x + threadIdx.x) >> 5;
  int lane = threadIdx.x & 31;
  if (node >= NN) return;            // wave-uniform exit
  float* row = msg + (long long)node * HF;
  float t[5];
  float s = 0.f, s2 = 0.f;
  int cnt = 0;
  for (int c = lane; c < HF; c += 32) {
    float v = row[c] + gb[c];
    t[cnt++] = v;
    s += v; s2 += v * v;
  }
  for (int o = 16; o >= 1; o >>= 1) {
    s  += __shfl_xor(s,  o, 32);
    s2 += __shfl_xor(s2, o, 32);
  }
  float mu  = s / (float)HF;
  float var = s2 / (float)HF - mu * mu;
  float rr  = rsqrtf(var + 1e-5f);
  cnt = 0;
  for (int c = lane; c < HF; c += 32)
    row[c] = (t[cnt++] - mu) * rr * gam[c] + bet[c];
}

// ---------------- GEMM2: hout = 2*hin + normed[N,152]@fW[152,128] + fb ------
// K padded 152 -> 160 (5 K-tiles). 256 threads = 8 waves = 8 N-tiles; block
// covers 32 node rows (2 M-tiles per wave).
__global__ __launch_bounds__(256) void k_gemm2(const float* __restrict__ nrm,
                                               const __bf16* __restrict__ Bpk,
                                               const float* __restrict__ fb,
                                               const float* __restrict__ hin,
                                               float* __restrict__ hout) {
  __shared__ __attribute__((aligned(32))) __bf16 Ap[2 * 5 * 32 * 16]; // 10 KB
  __shared__ __attribute__((aligned(32))) __bf16 Bp[8 * 5 * 32 * 16]; // 40 KB
  const int tid = threadIdx.x;
  const int m0  = blockIdx.x * 32;

  const unsigned ldsB = lds_off32(&Bp[0]);
  for (int c = tid; c < 2560; c += 256) {
    unsigned off = (unsigned)c * 16u;
    async_lds_b128(ldsB + off, off, Bpk);
  }
  for (int idx = tid; idx < 32 * 160; idx += 256) {
    int m32 = idx / 160, k = idx % 160;
    int node = m0 + m32;
    float val = (node < NN && k < HF) ? nrm[node * HF + k] : 0.0f;
    int mt = m32 >> 4, m = m32 & 15;
    int kt = k >> 5, r = k & 31;
    int g = r >> 4;
    int j = r & 15;          // A map
    Ap[(((mt * 5) + kt) * 32 + (g * 16 + m)) * 16 + j] = (__bf16)val;
  }
  wait_async0();
  __syncthreads();

  const int lane = tid & 31;
  const int wave = tid >> 5;
  const int g    = lane >> 4;
  const int l15  = lane & 15;

  v8f acc0 = {}, acc1 = {};
  for (int kt = 0; kt < 5; ++kt) {
    v16bf b  = *(const v16bf*)&Bp[((wave * 5 + kt) * 32 + lane) * 16];
    v16bf a0 = *(const v16bf*)&Ap[((0 * 5 + kt) * 32 + lane) * 16];
    v16bf a1 = *(const v16bf*)&Ap[((1 * 5 + kt) * 32 + lane) * 16];
    acc0 = __builtin_amdgcn_wmma_f32_16x16x32_bf16(false, a0, false, b,
                                                   (short)0, acc0, false, false);
    acc1 = __builtin_amdgcn_wmma_f32_16x16x32_bf16(false, a1, false, b,
                                                   (short)0, acc1, false, false);
  }
  const int n = wave * 16 + l15;
  const int row0 = m0 + g * 8;
  const float fbn = fb[n];
  if (m0 + 32 <= NN) {            // block-uniform fast path
#pragma unroll
    for (int r = 0; r < 8; ++r) {
      int node0 = row0 + r, node1 = row0 + r + 16;
      hout[node0 * HID + n] = 2.0f * hin[node0 * HID + n] + acc0[r] + fbn;
      hout[node1 * HID + n] = 2.0f * hin[node1 * HID + n] + acc1[r] + fbn;
    }
  } else {
#pragma unroll
    for (int r = 0; r < 8; ++r) {
      int node0 = row0 + r, node1 = row0 + r + 16;
      if (node0 < NN)
        hout[node0 * HID + n] = 2.0f * hin[node0 * HID + n] + acc0[r] + fbn;
      if (node1 < NN)
        hout[node1 * HID + n] = 2.0f * hin[node1 * HID + n] + acc1[r] + fbn;
    }
  }
}

// ---------------- launch ----------------
extern "C" void kernel_launch(void* const* d_in, const int* in_sizes, int n_in,
                              void* d_out, int out_size, void* d_ws, size_t ws_size,
                              hipStream_t stream) {
  (void)in_sizes; (void)n_in; (void)out_size; (void)ws_size;
  const int*   z     = (const int*)d_in[0];
  // d_in[1] = pos : dead in reference (edge_attr deleted) -> skipped
  const int*   ei    = (const int*)d_in[2];
  const float* emb   = (const float*)d_in[3];
  const float* gatW  = (const float*)d_in[4];
  const float* gatB  = (const float*)d_in[5];
  const float* a_src = (const float*)d_in[6];
  const float* a_dst = (const float*)d_in[7];
  const float* lng   = (const float*)d_in[8];
  const float* lnb   = (const float*)d_in[9];
  const float* ffW   = (const float*)d_in[10];
  const float* ffb   = (const float*)d_in[11];

  float* ws   = (float*)d_ws;
  float* h    = ws;                   // 6,400,000 f
  float* xp   = ws + 6400000;         // 8,000,000 f (padded stride 160)
  float* als  = ws + 14400000;        //   400,000 f
  float* ald  = ws + 14800000;        //   400,000 f
  float* ebuf = ws + 15200000;        // 6,800,000 f
  unsigned* maxb = (unsigned*)(ws + 22000000); // 400,000
  float* sumb = ws + 22400000;        //   400,000 f
  float* msg  = ws + 22800000;        // 7,600,000 f
  __bf16* bpk1 = (__bf16*)(ws + 30400000);     // 20480 bf16 (16B-aligned)
  __bf16* bpk2 = (__bf16*)(ws + 30410240);     // 20480 bf16
  float* out  = (float*)d_out;

  k_embed<<<(NN * HID + 255) / 256, 256, 0, stream>>>(z, emb, h);

  const int mt = (NN + 31) / 32;      // 1563 row tiles (32 rows per block)
  for (int b = 0; b < 3; ++b) {
    const float* W  = gatW + b * HID * HF;
    const float* gb = gatB + b * HF;
    const float* as = a_src + b * HEADS * OUTC;
    const float* ad = a_dst + b * HEADS * OUTC;
    const float* gm = lng + b * HF;
    const float* be = lnb + b * HF;
    const float* fW = ffW + b * HF * HID;
    const float* fb = ffb + b * HID;

    k_packB1<<<(HID * 160 + 255) / 256, 256, 0, stream>>>(W, bpk1);
    k_gemm1<<<mt, 320, 0, stream>>>(h, bpk1, xp);
    k_al<<<(NN * HEADS + 255) / 256, 256, 0, stream>>>(xp, as, ad, als, ald);
    // zero [maxb | sumb | msg] contiguous region: 8,400,000 floats
    k_zero<<<(8400000 + 255) / 256, 256, 0, stream>>>(ws + 22000000, 8400000);
    k_edge1<<<(EEDG * HEADS + 255) / 256, 256, 0, stream>>>(ei, als, ald, ebuf, maxb);
    k_edge2<<<(EEDG * HEADS + 255) / 256, 256, 0, stream>>>(ei, ebuf, maxb, sumb);
    k_alpha<<<(EEDG * HEADS + 255) / 256, 256, 0, stream>>>(ei, ebuf, sumb);
    long long e3 = (long long)EEDG * (HF / 4);
    k_edge3<<<(int)((e3 + 255) / 256), 256, 0, stream>>>(ei, ebuf, xp, msg);
    k_post<<<(NN * 32 + 255) / 256, 256, 0, stream>>>(msg, gb, gm, be);
    k_packB2<<<(160 * HID + 255) / 256, 256, 0, stream>>>(fW, bpk2);
    float* hout = (b == 2) ? out : h;
    k_gemm2<<<mt, 256, 0, stream>>>(msg, bpk2, fb, h, hout);
  }
}
